// AMK_Block_57002805952691
// MI455X (gfx1250) — compile-verified, matrix operand order checked by reading, not run on hardware
//
#include <hip/hip_runtime.h>

typedef __bf16 bf16;
typedef __bf16 bf16x16 __attribute__((ext_vector_type(16)));
typedef float  f32x8   __attribute__((ext_vector_type(8)));

#define B_     4
#define N_     2048
#define D_     768
#define DS_    64
#define INNER_ 3072
#define ROWS_  (B_ * N_)   // 8192

// ---------------------------------------------------------------------------
// B-operand pack layout: for K-block kb (32 rows) and column n we store the 32
// bf16 values in per-lane fragment order. Lane (n, hi) reads 32 contiguous
// bytes at ((kb/32 * N + n)*2 + hi)*16 halves. Adjacent 16-column blocks are
// +512 halves (+1KB) apart -> immediate offsets for register-blocked loads.
// ---------------------------------------------------------------------------
__device__ __forceinline__ size_t pack_index(int k, int n, int N) {
  int kb = k >> 5, kk = k & 31;
  int hi = (kk >> 3) & 1;                 // 0-7:0, 8-15:1, 16-23:0, 24-31:1
  int h  = (kk & 7) + ((kk >> 4) << 3);   // +8 for kk>=16
  return (((size_t)kb * N + n) * 2 + hi) * 16 + h;
}

// Register-blocked tile: one wave computes a 16 x (NB*16) strip. All NB B
// fragments are loaded into distinct registers BEFORE the WMMA group so the
// scheduler can clause the loads and stagger s_wait_loadcnt against the
// matrix pipe instead of round-tripping loadcnt==0 per WMMA.
template <int NB>
__device__ __forceinline__ void wmma_tile_nb(const bf16* __restrict__ A, int lda,
                                             const bf16* __restrict__ Bp, int N,
                                             int K, int m0, int n0, int lane,
                                             f32x8* acc) {
  const int nl = lane & 15, hi = lane >> 4;
  const bf16* arow = A + (size_t)(m0 + nl) * lda;
  const bf16* pb   = Bp + ((size_t)(n0 + nl) * 2 + hi) * 16;
  f32x8 zero = {};
#pragma unroll
  for (int j = 0; j < NB; ++j) acc[j] = zero;
  for (int kb = 0; kb < K; kb += 32) {
    bf16x16 af, bfs[NB];
    const bf16* pa = arow + kb + hi * 8;
    ((uint4*)&af)[0] = *(const uint4*)(pa);
    ((uint4*)&af)[1] = *(const uint4*)(pa + 16);
#pragma unroll
    for (int j = 0; j < NB; ++j) {
      ((uint4*)&bfs[j])[0] = ((const uint4*)(pb + j * 512))[0];
      ((uint4*)&bfs[j])[1] = ((const uint4*)(pb + j * 512))[1];
    }
    if (kb + 32 < K) {                      // prefetch next K-block
      __builtin_prefetch(arow + kb + 32, 0, 3);
      __builtin_prefetch(pb + (size_t)N * 32, 0, 3);
    }
#pragma unroll
    for (int j = 0; j < NB; ++j)
      acc[j] = __builtin_amdgcn_wmma_f32_16x16x32_bf16(false, af, false, bfs[j],
                                                       (short)0, acc[j], false,
                                                       false);
    pb += (size_t)N * 32;
  }
}

// Dual-B 2x2 variant for the gated-FFN up projection: one A fragment feeds
// 2 G-tiles + 2 U-tiles (4 WMMAs) per K-step; fragments loaded up front.
__device__ __forceinline__ void wmma_tile_gu(const bf16* __restrict__ A, int lda,
                                             const bf16* __restrict__ Bg,
                                             const bf16* __restrict__ Bu, int N,
                                             int K, int m0, int n0, int lane,
                                             f32x8* accg, f32x8* accu) {
  const int nl = lane & 15, hi = lane >> 4;
  const bf16* arow = A + (size_t)(m0 + nl) * lda;
  const bf16* pg   = Bg + ((size_t)(n0 + nl) * 2 + hi) * 16;
  const bf16* pu   = Bu + ((size_t)(n0 + nl) * 2 + hi) * 16;
  f32x8 zero = {};
  accg[0] = accg[1] = zero;
  accu[0] = accu[1] = zero;
  for (int kb = 0; kb < K; kb += 32) {
    bf16x16 af, bg[2], bu[2];
    const bf16* pa = arow + kb + hi * 8;
    ((uint4*)&af)[0] = *(const uint4*)(pa);
    ((uint4*)&af)[1] = *(const uint4*)(pa + 16);
#pragma unroll
    for (int j = 0; j < 2; ++j) {
      ((uint4*)&bg[j])[0] = ((const uint4*)(pg + j * 512))[0];
      ((uint4*)&bg[j])[1] = ((const uint4*)(pg + j * 512))[1];
      ((uint4*)&bu[j])[0] = ((const uint4*)(pu + j * 512))[0];
      ((uint4*)&bu[j])[1] = ((const uint4*)(pu + j * 512))[1];
    }
    if (kb + 32 < K) {
      __builtin_prefetch(arow + kb + 32, 0, 3);
      __builtin_prefetch(pg + (size_t)N * 32, 0, 3);
      __builtin_prefetch(pu + (size_t)N * 32, 0, 3);
    }
#pragma unroll
    for (int j = 0; j < 2; ++j) {
      accg[j] = __builtin_amdgcn_wmma_f32_16x16x32_bf16(false, af, false, bg[j],
                                                        (short)0, accg[j], false,
                                                        false);
      accu[j] = __builtin_amdgcn_wmma_f32_16x16x32_bf16(false, af, false, bu[j],
                                                        (short)0, accu[j], false,
                                                        false);
    }
    pg += (size_t)N * 32;
    pu += (size_t)N * 32;
  }
}

// ------------------------------- LN kernels --------------------------------
__global__ void k_ln1(const float* __restrict__ Qin, const float* __restrict__ X,
                      const float* __restrict__ g, const float* __restrict__ b,
                      float* __restrict__ Hf, bf16* __restrict__ Hb) {
  const int row = blockIdx.x;
  const float* x = Qin + (size_t)row * D_;
  __shared__ float red[256];
  const int tid = threadIdx.x;
  float s = 0.f;
  for (int i = tid; i < D_; i += 256) s += x[i];
  red[tid] = s; __syncthreads();
  for (int o = 128; o > 0; o >>= 1) { if (tid < o) red[tid] += red[tid + o]; __syncthreads(); }
  float mu = red[0] * (1.0f / D_);
  __syncthreads();
  float v = 0.f;
  for (int i = tid; i < D_; i += 256) { float d = x[i] - mu; v += d * d; }
  red[tid] = v; __syncthreads();
  for (int o = 128; o > 0; o >>= 1) { if (tid < o) red[tid] += red[tid + o]; __syncthreads(); }
  float rs = rsqrtf(red[0] * (1.0f / D_) + 1e-5f);
  for (int i = tid; i < D_; i += 256) {
    float h = (x[i] - mu) * rs * g[i] + b[i] + X[(size_t)row * D_ + i];
    Hf[(size_t)row * D_ + i] = h;
    Hb[(size_t)row * D_ + i] = (bf16)h;
  }
}

__global__ void k_ln2(const float* __restrict__ Qi, const float* __restrict__ g,
                      const float* __restrict__ b, bf16* __restrict__ out) {
  const int row = blockIdx.x;
  const float* x = Qi + (size_t)row * D_;
  __shared__ float red[256];
  const int tid = threadIdx.x;
  float s = 0.f;
  for (int i = tid; i < D_; i += 256) s += x[i];
  red[tid] = s; __syncthreads();
  for (int o = 128; o > 0; o >>= 1) { if (tid < o) red[tid] += red[tid + o]; __syncthreads(); }
  float mu = red[0] * (1.0f / D_);
  __syncthreads();
  float v = 0.f;
  for (int i = tid; i < D_; i += 256) { float d = x[i] - mu; v += d * d; }
  red[tid] = v; __syncthreads();
  for (int o = 128; o > 0; o >>= 1) { if (tid < o) red[tid] += red[tid + o]; __syncthreads(); }
  float rs = rsqrtf(red[0] * (1.0f / D_) + 1e-5f);
  for (int i = tid; i < D_; i += 256)
    out[(size_t)row * D_ + i] = (bf16)((x[i] - mu) * rs * g[i] + b[i]);
}

// --------------------------- pooling / hyper -------------------------------
__global__ void k_qpool(const float* __restrict__ Hf, float* __restrict__ qp) {
  int idx = blockIdx.x * 256 + threadIdx.x;       // (b, d)
  if (idx >= B_ * D_) return;
  int bb = idx / D_, dd = idx % D_;
  const float* p = Hf + (size_t)bb * N_ * D_ + dd;
  float s = 0.f;
  for (int n = 0; n < N_; ++n) s += p[(size_t)n * D_];
  qp[idx] = s * (1.0f / N_);
}

// Omega = q_pool @ hyper_w.T ; one wave per weight row, 4 batches per pass.
__global__ void k_omega(const float* __restrict__ qp, const float* __restrict__ hq,
                        const float* __restrict__ hk, float* __restrict__ omQ,
                        float* __restrict__ omK) {
  int wave = threadIdx.x >> 5, lane = threadIdx.x & 31;
  int row = blockIdx.x * 8 + wave;
  if (row >= D_ * DS_) return;
  const float* w = (blockIdx.y == 0 ? hq : hk) + (size_t)row * D_;
  float s0 = 0, s1 = 0, s2 = 0, s3 = 0;
  for (int i = lane; i < D_; i += 32) {
    float wv = w[i];
    s0 += wv * qp[i];
    s1 += wv * qp[D_ + i];
    s2 += wv * qp[2 * D_ + i];
    s3 += wv * qp[3 * D_ + i];
  }
  for (int o = 16; o > 0; o >>= 1) {
    s0 += __shfl_down(s0, o, 32);
    s1 += __shfl_down(s1, o, 32);
    s2 += __shfl_down(s2, o, 32);
    s3 += __shfl_down(s3, o, 32);
  }
  if (lane == 0) {
    float* dst = (blockIdx.y == 0 ? omQ : omK);
    dst[row]                = s0;
    dst[D_ * DS_ + row]     = s1;
    dst[2 * D_ * DS_ + row] = s2;
    dst[3 * D_ * DS_ + row] = s3;
  }
}

// ------------------------------ pack kernels -------------------------------
__global__ void k_pack_direct(const float* __restrict__ src, bf16* __restrict__ dst,
                              int K, int N, int batch, size_t srcStride,
                              size_t dstStride) {
  size_t idx = (size_t)blockIdx.x * 256 + threadIdx.x;
  size_t total = (size_t)batch * K * N;
  if (idx >= total) return;
  int n = (int)(idx % N);
  size_t t = idx / N;
  int k = (int)(t % K);
  int bb = (int)(t / K);
  float v = src[(size_t)bb * srcStride + (size_t)k * N + n];
  dst[(size_t)bb * dstStride + pack_index(k, n, N)] = (bf16)v;
}

__global__ void k_pack_transpose(const float* __restrict__ src, bf16* __restrict__ dst,
                                 int K, int N) {
  size_t idx = (size_t)blockIdx.x * 256 + threadIdx.x;
  if (idx >= (size_t)K * N) return;
  int n = (int)(idx % N);
  int k = (int)(idx / N);
  dst[pack_index(k, n, N)] = (bf16)src[(size_t)n * K + k];
}

// ------------------------------ GEMM kernels -------------------------------
// Phi = elu(H @ Omega + bias) + 1 ; one wave covers the full 64-wide N (NB=4).
__global__ void k_phi(const bf16* __restrict__ Hb, const bf16* __restrict__ OmP,
                      const float* __restrict__ bias, bf16* __restrict__ out,
                      int transposed) {
  const int bz = blockIdx.z;
  const bf16* A  = Hb  + (size_t)bz * N_ * D_;
  const bf16* Bp = OmP + (size_t)bz * D_ * DS_;
  int wave = threadIdx.x >> 5, lane = threadIdx.x & 31;
  int mt = blockIdx.x * 8 + wave;                 // 128 M-tiles
  if (mt >= N_ / 16) return;
  f32x8 acc[4];
  wmma_tile_nb<4>(A, D_, Bp, DS_, D_, mt * 16, 0, lane, acc);
  int nl = lane & 15, hi = lane >> 4;
#pragma unroll
  for (int j = 0; j < 4; ++j) {
    int n = j * 16 + nl;
    float bv = bias[n];
#pragma unroll
    for (int r = 0; r < 8; ++r) {
      int m = mt * 16 + r + 8 * hi;
      float v = acc[j][r] + bv;
      float phi = (v > 0.f) ? (v + 1.f) : __expf(v);   // elu(v)+1
      if (!transposed) out[((size_t)bz * N_ + m) * DS_ + n] = (bf16)phi;
      else             out[((size_t)bz * DS_ + n) * N_ + m] = (bf16)phi;
    }
  }
}

// C[b] = Phi_K^T @ H   (A = PhiKT [Ds,N], B = H packed [N, D]); NB=4.
__global__ void k_cmat(const bf16* __restrict__ PhiKT, const bf16* __restrict__ HP,
                       float* __restrict__ C) {
  const int bz = blockIdx.z;
  const bf16* A  = PhiKT + (size_t)bz * DS_ * N_;
  const bf16* Bp = HP    + (size_t)bz * N_ * D_;
  int wave = threadIdx.x >> 5, lane = threadIdx.x & 31;
  int tile = blockIdx.x * 8 + wave;
  const int NTB = D_ / 64;                        // 12
  if (tile >= (DS_ / 16) * NTB) return;
  int mt = tile / NTB, ntb = tile % NTB;
  f32x8 acc[4];
  wmma_tile_nb<4>(A, N_, Bp, D_, N_, mt * 16, ntb * 64, lane, acc);
  int nl = lane & 15, hi = lane >> 4;
#pragma unroll
  for (int j = 0; j < 4; ++j)
#pragma unroll
    for (int r = 0; r < 8; ++r) {
      int m = mt * 16 + r + 8 * hi;
      C[((size_t)bz * DS_ + m) * D_ + ntb * 64 + j * 16 + nl] = acc[j][r];
    }
}

__global__ void k_pks(const bf16* __restrict__ PhiKT, float* __restrict__ pks) {
  int idx = threadIdx.x;                          // B_*DS_ == 256
  if (idx >= B_ * DS_) return;
  const bf16* p = PhiKT + (size_t)idx * N_;
  float s = 0.f;
  for (int n = 0; n < N_; ++n) s += (float)p[n];
  pks[idx] = s;
}

__global__ void k_rnorm(const bf16* __restrict__ PhiQ, const float* __restrict__ pks,
                        float* __restrict__ rn) {
  int idx = blockIdx.x * 256 + threadIdx.x;       // (b, n)
  if (idx >= ROWS_) return;
  int bb = idx / N_;
  const bf16* p = PhiQ + (size_t)idx * DS_;
  const float* ks = pks + bb * DS_;
  float s = 0.f;
  for (int i = 0; i < DS_; ++i) s += (float)p[i] * ks[i];
  rn[idx] = 1.0f / (fabsf(s) + 1.0f);
}

// m = (Phi_Q @ C) / Norm - H ; NB=4.
__global__ void k_attr(const bf16* __restrict__ PhiQ, const bf16* __restrict__ CP,
                       const float* __restrict__ rn, const float* __restrict__ Hf,
                       bf16* __restrict__ m_out) {
  const int bz = blockIdx.z;
  const bf16* A  = PhiQ + (size_t)bz * N_ * DS_;
  const bf16* Bp = CP   + (size_t)bz * DS_ * D_;
  int wave = threadIdx.x >> 5, lane = threadIdx.x & 31;
  int tile = blockIdx.x * 8 + wave;
  const int NTB = D_ / 64;                        // 12
  if (tile >= (N_ / 16) * NTB) return;
  int mt = tile / NTB, ntb = tile % NTB;
  f32x8 acc[4];
  wmma_tile_nb<4>(A, DS_, Bp, D_, DS_, mt * 16, ntb * 64, lane, acc);
  int nl = lane & 15, hi = lane >> 4;
#pragma unroll
  for (int j = 0; j < 4; ++j)
#pragma unroll
    for (int r = 0; r < 8; ++r) {
      int m = mt * 16 + r + 8 * hi;
      size_t row = (size_t)bz * N_ + m;
      int col = ntb * 64 + j * 16 + nl;
      float v = acc[j][r] * rn[row] - Hf[row * D_ + col];
      m_out[row * D_ + col] = (bf16)v;
    }
}

// Q_interact = Q_in + softplus(dt) * (m @ Wm.T + mb) ; NB=4.
__global__ void k_mproj(const bf16* __restrict__ M, const bf16* __restrict__ WmP,
                        const float* __restrict__ mb, const float* __restrict__ dt,
                        const float* __restrict__ Qin, float* __restrict__ Qi) {
  int wave = threadIdx.x >> 5, lane = threadIdx.x & 31;
  int tile = blockIdx.x * 8 + wave;
  const int NTB = D_ / 64;                        // 12
  if (tile >= (ROWS_ / 16) * NTB) return;
  int mt = tile / NTB, ntb = tile % NTB;
  f32x8 acc[4];
  wmma_tile_nb<4>(M, D_, WmP, D_, D_, mt * 16, ntb * 64, lane, acc);
  float sp = log1pf(__expf(dt[0]));
  int nl = lane & 15, hi = lane >> 4;
#pragma unroll
  for (int j = 0; j < 4; ++j)
#pragma unroll
    for (int r = 0; r < 8; ++r) {
      int m = mt * 16 + r + 8 * hi;
      int col = ntb * 64 + j * 16 + nl;
      size_t o = (size_t)m * D_ + col;
      Qi[o] = Qin[o] + sp * (acc[j][r] + mb[col]);
    }
}

// H_ffn = silu(Qn2 @ Wg.T) * (Qn2 @ Wu.T) ; 2 G-tiles + 2 U-tiles per wave.
__global__ void k_gu(const bf16* __restrict__ Qn2, const bf16* __restrict__ WgP,
                     const bf16* __restrict__ WuP, bf16* __restrict__ hffn) {
  int wave = threadIdx.x >> 5, lane = threadIdx.x & 31;
  int tile = blockIdx.x * 8 + wave;
  const int NTB = INNER_ / 32;                    // 96
  if (tile >= (ROWS_ / 16) * NTB) return;
  int mt = tile / NTB, ntb = tile % NTB;
  f32x8 g[2], u[2];
  wmma_tile_gu(Qn2, D_, WgP, WuP, INNER_, D_, mt * 16, ntb * 32, lane, g, u);
  int nl = lane & 15, hi = lane >> 4;
#pragma unroll
  for (int j = 0; j < 2; ++j)
#pragma unroll
    for (int r = 0; r < 8; ++r) {
      int m = mt * 16 + r + 8 * hi;
      float gv = g[j][r];
      float h = (gv / (1.f + __expf(-gv))) * u[j][r];   // silu(g)*u
      hffn[(size_t)m * INNER_ + ntb * 32 + j * 16 + nl] = (bf16)h;
    }
}

// Depthwise conv over sequence dim, K=3, pad 1.
__global__ void k_conv(const bf16* __restrict__ hffn, const float* __restrict__ dw,
                       bf16* __restrict__ out) {
  size_t idx = (size_t)blockIdx.x * 256 + threadIdx.x;
  if (idx >= (size_t)ROWS_ * INNER_) return;
  int c = (int)(idx % INNER_);
  size_t rn = idx / INNER_;
  int n = (int)(rn % N_);
  float w0 = dw[c * 3], w1 = dw[c * 3 + 1], w2 = dw[c * 3 + 2];
  float x0 = (float)hffn[idx];
  float xm = (n > 0)      ? (float)hffn[idx - INNER_] : 0.f;
  float xp = (n < N_ - 1) ? (float)hffn[idx + INNER_] : 0.f;
  out[idx] = (bf16)(w0 * xm + w1 * x0 + w2 * xp);
}

// out = Q_interact + conv @ Wdn.T ; NB=4.
__global__ void k_down(const bf16* __restrict__ cv, const bf16* __restrict__ WdP,
                       const float* __restrict__ Qi, float* __restrict__ out) {
  int wave = threadIdx.x >> 5, lane = threadIdx.x & 31;
  int tile = blockIdx.x * 8 + wave;
  const int NTB = D_ / 64;                        // 12
  if (tile >= (ROWS_ / 16) * NTB) return;
  int mt = tile / NTB, ntb = tile % NTB;
  f32x8 acc[4];
  wmma_tile_nb<4>(cv, INNER_, WdP, D_, INNER_, mt * 16, ntb * 64, lane, acc);
  int nl = lane & 15, hi = lane >> 4;
#pragma unroll
  for (int j = 0; j < 4; ++j)
#pragma unroll
    for (int r = 0; r < 8; ++r) {
      int m = mt * 16 + r + 8 * hi;
      size_t o = (size_t)m * D_ + ntb * 64 + j * 16 + nl;
      out[o] = Qi[o] + acc[j][r];
    }
}

// ------------------------------- launcher ----------------------------------
extern "C" void kernel_launch(void* const* d_in, const int* in_sizes, int n_in,
                              void* d_out, int out_size, void* d_ws, size_t ws_size,
                              hipStream_t stream) {
  (void)in_sizes; (void)n_in; (void)out_size; (void)ws_size;
  const float* Qin = (const float*)d_in[0];
  const float* X   = (const float*)d_in[1];
  const float* dt  = (const float*)d_in[2];
  const float* hq  = (const float*)d_in[3];
  const float* hk  = (const float*)d_in[4];
  const float* BQ  = (const float*)d_in[5];
  const float* BK  = (const float*)d_in[6];
  const float* Wm  = (const float*)d_in[7];
  const float* mb  = (const float*)d_in[8];
  const float* n1g = (const float*)d_in[9];
  const float* n1b = (const float*)d_in[10];
  const float* n2g = (const float*)d_in[11];
  const float* n2b = (const float*)d_in[12];
  const float* Wup = (const float*)d_in[13];
  const float* dw  = (const float*)d_in[14];
  const float* Wdn = (const float*)d_in[15];
  float* out = (float*)d_out;

  char* base = (char*)d_ws;
  size_t off = 0;
  auto alloc = [&](size_t bytes) -> void* {
    void* p = base + off;
    off = (off + bytes + 255) & ~(size_t)255;
    return p;
  };
  float* Hf    = (float*)alloc((size_t)ROWS_ * D_ * 4);
  bf16*  Hb    = (bf16*) alloc((size_t)ROWS_ * D_ * 2);
  float* qp    = (float*)alloc((size_t)B_ * D_ * 4);
  float* omQ   = (float*)alloc((size_t)B_ * D_ * DS_ * 4);
  float* omK   = (float*)alloc((size_t)B_ * D_ * DS_ * 4);
  bf16*  omQP  = (bf16*) alloc((size_t)B_ * D_ * DS_ * 2);
  bf16*  omKP  = (bf16*) alloc((size_t)B_ * D_ * DS_ * 2);
  bf16*  phiQ  = (bf16*) alloc((size_t)B_ * N_ * DS_ * 2);
  bf16*  phiKT = (bf16*) alloc((size_t)B_ * DS_ * N_ * 2);
  bf16*  HP    = (bf16*) alloc((size_t)B_ * N_ * D_ * 2);
  float* Cf    = (float*)alloc((size_t)B_ * DS_ * D_ * 4);
  bf16*  CP    = (bf16*) alloc((size_t)B_ * DS_ * D_ * 2);
  float* pks   = (float*)alloc((size_t)B_ * DS_ * 4);
  float* rn    = (float*)alloc((size_t)ROWS_ * 4);
  bf16*  mbuf  = (bf16*) alloc((size_t)ROWS_ * D_ * 2);
  float* Qi    = (float*)alloc((size_t)ROWS_ * D_ * 4);
  bf16*  Qn2   = (bf16*) alloc((size_t)ROWS_ * D_ * 2);
  bf16*  WmP   = (bf16*) alloc((size_t)D_ * D_ * 2);
  bf16*  WgP   = (bf16*) alloc((size_t)D_ * INNER_ * 2);
  bf16*  WuP   = (bf16*) alloc((size_t)D_ * INNER_ * 2);
  bf16*  WdP   = (bf16*) alloc((size_t)INNER_ * D_ * 2);
  bf16*  hffn  = (bf16*) alloc((size_t)ROWS_ * INNER_ * 2);
  bf16*  cv    = (bf16*) alloc((size_t)ROWS_ * INNER_ * 2);

  // Weight packs (f32 -> bf16 fragment layout), independent of activations.
  k_pack_transpose<<<(D_ * D_ + 255) / 256, 256, 0, stream>>>(Wm, WmP, D_, D_);
  k_pack_transpose<<<(D_ * INNER_ + 255) / 256, 256, 0, stream>>>(Wup, WgP, D_, INNER_);
  k_pack_transpose<<<(D_ * INNER_ + 255) / 256, 256, 0, stream>>>(
      Wup + (size_t)INNER_ * D_, WuP, D_, INNER_);
  k_pack_transpose<<<(INNER_ * D_ + 255) / 256, 256, 0, stream>>>(Wdn, WdP, INNER_, D_);

  // H = LN(Q_in) + X
  k_ln1<<<ROWS_, 256, 0, stream>>>(Qin, X, n1g, n1b, Hf, Hb);
  // q_pool = mean_n H
  k_qpool<<<(B_ * D_ + 255) / 256, 256, 0, stream>>>(Hf, qp);
  // Omega_Q/K GEMV
  k_omega<<<dim3((D_ * DS_) / 8, 2), 256, 0, stream>>>(qp, hq, hk, omQ, omK);
  k_pack_direct<<<((size_t)B_ * D_ * DS_ + 255) / 256, 256, 0, stream>>>(
      omQ, omQP, D_, DS_, B_, (size_t)D_ * DS_, (size_t)D_ * DS_);
  k_pack_direct<<<((size_t)B_ * D_ * DS_ + 255) / 256, 256, 0, stream>>>(
      omK, omKP, D_, DS_, B_, (size_t)D_ * DS_, (size_t)D_ * DS_);
  // Phi_Q [b,n,Ds], Phi_K stored transposed [b,Ds,n]
  k_phi<<<dim3(16, 1, B_), 256, 0, stream>>>(Hb, omQP, BQ, phiQ, 0);
  k_phi<<<dim3(16, 1, B_), 256, 0, stream>>>(Hb, omKP, BK, phiKT, 1);
  // H as B-operand of C-GEMM
  k_pack_direct<<<((size_t)B_ * N_ * D_ + 255) / 256, 256, 0, stream>>>(
      Hf, HP, N_, D_, B_, (size_t)N_ * D_, (size_t)N_ * D_);
  // C = Phi_K^T @ H
  k_cmat<<<dim3(6, 1, B_), 256, 0, stream>>>(phiKT, HP, Cf);
  k_pack_direct<<<((size_t)B_ * DS_ * D_ + 255) / 256, 256, 0, stream>>>(
      Cf, CP, DS_, D_, B_, (size_t)DS_ * D_, (size_t)DS_ * D_);
  // phi_k_sum and 1/Norm
  k_pks<<<1, 256, 0, stream>>>(phiKT, pks);
  k_rnorm<<<ROWS_ / 256, 256, 0, stream>>>(phiQ, pks, rn);
  // m = Attraction/Norm - H
  k_attr<<<dim3(192, 1, B_), 256, 0, stream>>>(phiQ, CP, rn, Hf, mbuf);
  // Q_interact
  k_mproj<<<768, 256, 0, stream>>>(mbuf, WmP, mb, dt, Qin, Qi);
  // LN2
  k_ln2<<<ROWS_, 256, 0, stream>>>(Qi, n2g, n2b, Qn2);
  // gated FFN up
  k_gu<<<6144, 256, 0, stream>>>(Qn2, WgP, WuP, hffn);
  // depthwise conv
  k_conv<<<(int)(((size_t)ROWS_ * INNER_ + 255) / 256), 256, 0, stream>>>(hffn, dw, cv);
  // FFN down + residual
  k_down<<<768, 256, 0, stream>>>(cv, WdP, Qi, out);
}
